// Net_35321811042895
// MI455X (gfx1250) — matrix operand order repaired and checked by reading, not compiled, not run on hardware
//
#include <hip/hip_runtime.h>
#include <hip/hip_bf16.h>
#include <math.h>

#define E_EDGES 131072
#define NBATCH  8

typedef __bf16 bf16;
typedef __attribute__((ext_vector_type(16))) __bf16 v16bf;
typedef __attribute__((ext_vector_type(8)))  __bf16 v8bf;
typedef __attribute__((ext_vector_type(8)))  float  v8f;

// ---------------- device helpers ----------------
__device__ __forceinline__ bf16 f2bf(float f) {
    unsigned u = __builtin_bit_cast(unsigned, f);
    unsigned r = u + 0x7FFFu + ((u >> 16) & 1u);   // RNE
    unsigned short h = (unsigned short)(r >> 16);
    return __builtin_bit_cast(bf16, h);
}
__device__ __forceinline__ float eluf(float x) { return x > 0.f ? x : expm1f(x); }
__device__ __forceinline__ unsigned fenc(float f) {   // order-preserving float->uint
    unsigned u = __builtin_bit_cast(unsigned, f);
    return (u & 0x80000000u) ? ~u : (u | 0x80000000u);
}
__device__ __forceinline__ float fdec(unsigned e) {
    unsigned u = (e & 0x80000000u) ? (e ^ 0x80000000u) : ~e;
    return __builtin_bit_cast(float, u);
}

// ---------------- elementwise / fill ----------------
__global__ void k_fill_f(float* p, float v, int n) {
    int i = blockIdx.x * blockDim.x + threadIdx.x; if (i < n) p[i] = v;
}
__global__ void k_fill_i(int* p, int v, int n) {
    int i = blockIdx.x * blockDim.x + threadIdx.x; if (i < n) p[i] = v;
}
__global__ void k_cvt_bf(const float* __restrict__ in, bf16* __restrict__ out, int n) {
    int i = blockIdx.x * blockDim.x + threadIdx.x; if (i < n) out[i] = f2bf(in[i]);
}
// batched convert fp32 [B][K][N] -> bf16 transposed [B][N][K]; K and K*N are powers of two
__global__ void k_cvt_bf_t(const float* __restrict__ in, bf16* __restrict__ out,
                           int n, int kshift, int knshift, int N) {
    int idx = blockIdx.x * blockDim.x + threadIdx.x; if (idx >= n) return;
    int b = idx >> knshift;
    int r = idx & ((1 << knshift) - 1);
    int col = r >> kshift;              // n (0..N-1)
    int k = r & ((1 << kshift) - 1);    // k
    out[idx] = f2bf(in[((long)b << knshift) + ((long)k * N) + col]);
}

// ---- WMMA GEMM: C[M,N] += A[M,K](bf16,row-major) * BT[N,K](bf16) ----
// 4 waves/block; each wave owns a 16x32 C tile (two 16x16 tiles sharing one A
// fragment -> 2 independent WMMA chains per A load). K multiple of 32.
// Edge tiles use clamped loads (never stored) + guarded stores: EXEC full at WMMA.
__global__ void k_gemm_bf16(const bf16* __restrict__ A, const bf16* __restrict__ BT,
                            float* __restrict__ C, int M, int N, int K,
                            int lda, int ldc) {
    const int lane = threadIdx.x & 31;
    const int wave = threadIdx.x >> 5;
    const int half = lane >> 4;
    const int l15  = lane & 15;
    const int m0 = blockIdx.y * 16;
    const int n0 = (blockIdx.x * 8 + wave * 2) * 16;
    if (n0 >= N) return;                           // wave-uniform skip
    const int am  = min(m0 + l15, M - 1);
    const int bn0 = min(n0 + l15, N - 1);
    const int bn1 = min(n0 + 16 + l15, N - 1);
    const bf16* ap  = A + (long)am * lda;
    const bf16* bp0 = BT + (long)bn0 * K;
    const bf16* bp1 = BT + (long)bn1 * K;

    v8f acc0, acc1;
#pragma unroll
    for (int v = 0; v < 8; ++v) {
        int mr = min(m0 + v + 8 * half, M - 1);
        acc0[v] = C[(long)mr * ldc + bn0];
        acc1[v] = C[(long)mr * ldc + bn1];
    }
    for (int k0 = 0; k0 < K; k0 += 32) {
        // A lane layout: k = k0 + e + 16*half (contiguous, matches B)
        v8bf a0 = *(const v8bf*)(ap + k0 + 16 * half);
        v8bf a1 = *(const v8bf*)(ap + k0 + 16 * half + 8);
        // B lane layout: k = k0 + e + 16*half, contiguous in transposed BT
        v16bf b0 = *(const v16bf*)(bp0 + k0 + 16 * half);
        v16bf b1 = *(const v16bf*)(bp1 + k0 + 16 * half);
        v16bf a = __builtin_shufflevector(a0, a1, 0, 1, 2, 3, 4, 5, 6, 7,
                                          8, 9, 10, 11, 12, 13, 14, 15);
        acc0 = __builtin_amdgcn_wmma_f32_16x16x32_bf16(false, a, false, b0,
                                                       (short)0, acc0, false, false);
        acc1 = __builtin_amdgcn_wmma_f32_16x16x32_bf16(false, a, false, b1,
                                                       (short)0, acc1, false, false);
    }
    const int nc0 = n0 + l15;
    const int nc1 = n0 + 16 + l15;
#pragma unroll
    for (int v = 0; v < 8; ++v) {
        int mr = m0 + v + 8 * half;
        if (mr < M) {
            if (nc0 < N) C[(long)mr * ldc + nc0] = acc0[v];
            if (nc1 < N) C[(long)mr * ldc + nc1] = acc1[v];
        }
    }
}

// ---------------- spline basis (K=5, degree-1 open B-spline, dim 2) ----------------
__global__ void k_basis5(const float* __restrict__ attr, const float* __restrict__ ev,
                         float* __restrict__ bas, int* __restrict__ wi, int E) {
    int e = blockIdx.x * blockDim.x + threadIdx.x; if (e >= E) return;
    float v0 = attr[2 * e] * 4.f, v1 = attr[2 * e + 1] * 4.f;
    float fl0 = floorf(v0), fl1 = floorf(v1);
    int lo0 = min(max((int)fl0, 0), 4), lo1 = min(max((int)fl1, 0), 4);
    int hi0 = min(lo0 + 1, 4), hi1 = min(lo1 + 1, 4);
    float f0 = v0 - fl0, f1 = v1 - fl1;
    float m = ev[e];
    bas[4 * e + 0] = (1.f - f0) * (1.f - f1) * m; wi[4 * e + 0] = lo0 + 5 * lo1;
    bas[4 * e + 1] = (1.f - f0) * f1 * m;         wi[4 * e + 1] = lo0 + 5 * hi1;
    bas[4 * e + 2] = f0 * (1.f - f1) * m;         wi[4 * e + 2] = hi0 + 5 * lo1;
    bas[4 * e + 3] = f0 * f1 * m;                 wi[4 * e + 3] = hi0 + 5 * hi1;
}

__global__ void k_deg(const int* __restrict__ dst, const float* __restrict__ ev,
                      float* __restrict__ deg, int E) {
    int e = blockIdx.x * blockDim.x + threadIdx.x;
    if (e < E) atomicAdd(&deg[dst[e]], ev[e]);
}

// scatter for one spline kernel index k (C = 1<<cs channels)
__global__ void k_scatter_tap(const float* __restrict__ x, const int* __restrict__ src,
                              const int* __restrict__ dst, const float* __restrict__ bas,
                              const int* __restrict__ wi, float* __restrict__ buf,
                              int total, int cs, int k) {
    int idx = blockIdx.x * blockDim.x + threadIdx.x;
    if (idx >= total) return;
    int e = idx >> cs, c = idx & ((1 << cs) - 1);
    float w = 0.f;
#pragma unroll
    for (int s = 0; s < 4; ++s) if (wi[4 * e + s] == k) w += bas[4 * e + s];
    if (w != 0.f)
        atomicAdd(&buf[((long)dst[e] << cs) + c], w * x[((long)src[e] << cs) + c]);
}

// K=1 spline conv reduces to masked neighbor sum
__global__ void k_scatter_mean(const float* __restrict__ x, const int* __restrict__ src,
                               const int* __restrict__ dst, const float* __restrict__ ev,
                               float* __restrict__ buf, int total, int cs) {
    int idx = blockIdx.x * blockDim.x + threadIdx.x;
    if (idx >= total) return;
    int e = idx >> cs, c = idx & ((1 << cs) - 1);
    float w = ev[e];
    if (w != 0.f)
        atomicAdd(&buf[((long)dst[e] << cs) + c], w * x[((long)src[e] << cs) + c]);
}

__global__ void k_scale_deg(float* __restrict__ a, const float* __restrict__ deg,
                            int total, int cs) {
    int idx = blockIdx.x * blockDim.x + threadIdx.x;
    if (idx < total) a[idx] /= fmaxf(deg[idx >> cs], 1.f);
}

// io = [elu]( (io + bias) * g/sqrt(1+eps) + bb )
__global__ void k_bn_ep(float* __restrict__ io, const float* __restrict__ bias,
                        const float* __restrict__ g, const float* __restrict__ bb,
                        int total, int cs, int doElu) {
    int idx = blockIdx.x * blockDim.x + threadIdx.x;
    if (idx >= total) return;
    int c = idx & ((1 << cs) - 1);
    float t = io[idx] + bias[c];
    t = t * (g[c] * rsqrtf(1.f + 1e-5f)) + bb[c];
    io[idx] = doElu ? eluf(t) : t;
}

__global__ void k_add_elu(float* __restrict__ out, const float* __restrict__ a,
                          const float* __restrict__ b, int n) {
    int i = blockIdx.x * blockDim.x + threadIdx.x;
    if (i < n) out[i] = eluf(a[i] + b[i]);
}

// ---------------- conv1 (Cin=1) special path ----------------
__global__ void k_conv1_scatter(const float* __restrict__ x, const int* __restrict__ src,
                                const int* __restrict__ dst, const float* __restrict__ bas,
                                const int* __restrict__ wi, float* __restrict__ buf, int E) {
    int e = blockIdx.x * blockDim.x + threadIdx.x; if (e >= E) return;
    float xv = x[src[e]];
    int d = dst[e];
#pragma unroll
    for (int s = 0; s < 4; ++s) atomicAdd(&buf[d * 25 + wi[4 * e + s]], bas[4 * e + s] * xv);
}
__global__ void k_conv1_finish(const float* __restrict__ buf, const float* __restrict__ x,
                               const float* __restrict__ W, const float* __restrict__ root,
                               const float* __restrict__ b, const float* __restrict__ g,
                               const float* __restrict__ bb, const float* __restrict__ deg,
                               float* __restrict__ out, int total) {
    int idx = blockIdx.x * blockDim.x + threadIdx.x;
    if (idx >= total) return;
    int n = idx >> 6, o = idx & 63;
    float acc = 0.f;
#pragma unroll
    for (int k = 0; k < 25; ++k) acc += buf[n * 25 + k] * W[k * 64 + o];
    acc = acc / fmaxf(deg[n], 1.f) + x[n] * root[o] + b[o];
    acc = acc * (g[o] * rsqrtf(1.f + 1e-5f)) + bb[o];
    out[idx] = eluf(acc);
}

// ---------------- pooling ----------------
__global__ void k_cluster(const float* __restrict__ pos, int* __restrict__ cl, int N,
                          int bdiv, float sz, int nx, int ny, int slotMul) {
    int n = blockIdx.x * blockDim.x + threadIdx.x; if (n >= N) return;
    int cx = min(max((int)floorf(pos[2 * n] / sz), 0), nx - 1);
    int cy = min(max((int)floorf(pos[2 * n + 1] / sz), 0), ny - 1);
    cl[n] = (n / bdiv) * slotMul + cy * nx + cx;
}
__global__ void k_pool_feat(const float* __restrict__ h, const float* __restrict__ vn,
                            const int* __restrict__ cl, unsigned* __restrict__ xenc,
                            int total, int cs) {
    int idx = blockIdx.x * blockDim.x + threadIdx.x;
    if (idx >= total) return;
    int n = idx >> cs;
    if (vn[n] > 0.f)
        atomicMax(&xenc[((long)cl[n] << cs) + (idx & ((1 << cs) - 1))], fenc(h[idx]));
}
__global__ void k_pool_node(const float* __restrict__ pos, const float* __restrict__ vn,
                            const int* __restrict__ cl, float* __restrict__ cnt,
                            float* __restrict__ psum, int N) {
    int n = blockIdx.x * blockDim.x + threadIdx.x; if (n >= N) return;
    if (vn[n] > 0.f) {
        int c = cl[n];
        atomicAdd(&cnt[c], 1.f);
        atomicAdd(&psum[2 * c], pos[2 * n]);
        atomicAdd(&psum[2 * c + 1], pos[2 * n + 1]);
    }
}
__global__ void k_pool_fin_feat(const unsigned* __restrict__ xenc, const float* __restrict__ cnt,
                                float* __restrict__ xn, int total, int cs) {
    int idx = blockIdx.x * blockDim.x + threadIdx.x;
    if (idx < total) xn[idx] = (cnt[idx >> cs] > 0.f) ? fdec(xenc[idx]) : 0.f;
}
__global__ void k_pool_fin_node(const float* __restrict__ cnt, const float* __restrict__ psum,
                                float* __restrict__ pn, float* __restrict__ vn, int Nn) {
    int n = blockIdx.x * blockDim.x + threadIdx.x; if (n >= Nn) return;
    float c = cnt[n];
    pn[2 * n] = psum[2 * n] / fmaxf(c, 1.f);
    pn[2 * n + 1] = psum[2 * n + 1] / fmaxf(c, 1.f);
    vn[n] = c > 0.f ? 1.f : 0.f;
}
// dedup duplicate coarse edges with a CAS hash table (equiv. to argsort+first-occurrence)
__global__ void k_pool_edges(const int* __restrict__ srcO, const int* __restrict__ dstO,
                             const float* __restrict__ evO, const int* __restrict__ cl,
                             int* __restrict__ srcN, int* __restrict__ dstN,
                             float* __restrict__ evN, int* __restrict__ hash,
                             unsigned hmask, int Nn, int E) {
    int e = blockIdx.x * blockDim.x + threadIdx.x; if (e >= E) return;
    int sm = cl[srcO[e]], dm = cl[dstO[e]];
    srcN[e] = sm; dstN[e] = dm;
    float ok = 0.f;
    if (evO[e] > 0.f && sm != dm) {
        int key = sm * Nn + dm;
        unsigned h = ((unsigned)key * 2654435761u) & hmask;
        for (;;) {
            int prev = atomicCAS(&hash[h], -1, key);
            if (prev == -1) { ok = 1.f; break; }
            if (prev == key) break;
            h = (h + 1) & hmask;
        }
    }
    evN[e] = ok;
}
__global__ void k_pool_cart(const int* __restrict__ srcN, const int* __restrict__ dstN,
                            const float* __restrict__ evN, const float* __restrict__ pn,
                            float* __restrict__ attrN, unsigned* __restrict__ menc, int E) {
    int e = blockIdx.x * blockDim.x + threadIdx.x; if (e >= E) return;
    if (evN[e] > 0.f) {
        float c0 = pn[2 * dstN[e]] - pn[2 * srcN[e]];
        float c1 = pn[2 * dstN[e] + 1] - pn[2 * srcN[e] + 1];
        attrN[2 * e] = c0; attrN[2 * e + 1] = c1;
        float am = fmaxf(fabsf(c0), fabsf(c1));   // >= 0: raw bits are order-monotonic
        atomicMax(menc, __builtin_bit_cast(unsigned, am));
    } else {
        attrN[2 * e] = 0.f; attrN[2 * e + 1] = 0.f;
    }
}
__global__ void k_pool_attr(float* __restrict__ attrN, const float* __restrict__ evN,
                            const unsigned* __restrict__ menc, int E) {
    int e = blockIdx.x * blockDim.x + threadIdx.x; if (e >= E) return;
    float m = fmaxf(__builtin_bit_cast(float, *menc), 1e-12f);
    if (evN[e] > 0.f) {
        attrN[2 * e]     = attrN[2 * e] / (2.f * m) + 0.5f;
        attrN[2 * e + 1] = attrN[2 * e + 1] / (2.f * m) + 0.5f;
    }
}

// ---------------- FC tail ----------------
__global__ void k_bias_elu(float* __restrict__ io, const float* __restrict__ bias,
                           int total, int cs) {
    int idx = blockIdx.x * blockDim.x + threadIdx.x;
    if (idx < total) io[idx] = eluf(io[idx] + bias[idx & ((1 << cs) - 1)]);
}
__global__ void k_logsoftmax(const float* __restrict__ z, const float* __restrict__ bias,
                             float* __restrict__ out, int C) {
    __shared__ float red[128];
    int r = blockIdx.x, t = threadIdx.x;
    float mx = -1e30f;
    for (int c = t; c < C; c += 128) mx = fmaxf(mx, z[r * C + c] + bias[c]);
    red[t] = mx; __syncthreads();
    for (int s = 64; s > 0; s >>= 1) { if (t < s) red[t] = fmaxf(red[t], red[t + s]); __syncthreads(); }
    mx = red[0]; __syncthreads();
    float sm = 0.f;
    for (int c = t; c < C; c += 128) sm += expf(z[r * C + c] + bias[c] - mx);
    red[t] = sm; __syncthreads();
    for (int s = 64; s > 0; s >>= 1) { if (t < s) red[t] += red[t + s]; __syncthreads(); }
    float lse = logf(red[0]) + mx;
    for (int c = t; c < C; c += 128) out[r * C + c] = z[r * C + c] + bias[c] - lse;
}

// ================= host side =================
namespace {

struct WS {
    float *F0, *F1, *F2, *F3, *F4, *BUF;
    bf16  *HX, *HB, *WH;
    float *BAS; int *WI;
    float *DEG, *VNA, *VNB, *POSA, *POSB;
    int   *CL, *SRCA, *DSTA, *SRCB, *DSTB;
    float *EVA, *EVB, *ATTRA, *ATTRB;
    unsigned *XENC; float *CNT, *PSUM;
    int *HASH; unsigned *MENC;
};

inline int gup(long n) { return (int)((n + 255) / 256); }
inline int ilog2(int v) { int s = 0; while ((1 << s) < v) ++s; return s; }

inline void fillf(float* p, float v, long n, hipStream_t s) {
    k_fill_f<<<gup(n), 256, 0, s>>>(p, v, (int)n);
}
inline void filli(int* p, int v, long n, hipStream_t s) {
    k_fill_i<<<gup(n), 256, 0, s>>>(p, v, (int)n);
}
inline void cvtbf(const float* in, bf16* out, long n, hipStream_t s) {
    k_cvt_bf<<<gup(n), 256, 0, s>>>(in, out, (int)n);
}
// convert+transpose [B][K][N] fp32 -> [B][N][K] bf16 (K, K*N powers of two)
inline void cvtbf_t(const float* in, bf16* out, int B, int K, int N, hipStream_t s) {
    long n = (long)B * K * N;
    k_cvt_bf_t<<<gup(n), 256, 0, s>>>(in, out, (int)n, ilog2(K), ilog2(K * N), N);
}
inline void gemm(const bf16* A, const bf16* BT, float* C, int M, int N, int K,
                 int lda, int ldc, hipStream_t s) {
    dim3 g((N + 127) / 128, (M + 15) / 16);
    k_gemm_bf16<<<g, 128, 0, s>>>(A, BT, C, M, N, K, lda, ldc);
}

// SplineConv K=5 with mean aggregation + root + bias + BN (+ELU)
void conv5(const float* xin, float* out, const float* W, const float* root,
           const float* bias, const float* g, const float* bb, int doElu,
           int N, int Cin, int Cout, const int* src, const int* dst,
           WS& w, hipStream_t s) {
    const int cs = ilog2(Cin), co = ilog2(Cout);
    cvtbf(xin, w.HX, (long)N * Cin, s);
    cvtbf_t(W, w.WH, 25, Cin, Cout, s);                    // 25 x [Cout][Cin] (transposed)
    fillf(out, 0.f, (long)N * Cout, s);
    for (int k = 0; k < 25; ++k) {
        fillf(w.BUF, 0.f, (long)N * Cin, s);
        int t = E_EDGES << cs;
        k_scatter_tap<<<gup(t), 256, 0, s>>>(xin, src, dst, w.BAS, w.WI, w.BUF, t, cs, k);
        cvtbf(w.BUF, w.HB, (long)N * Cin, s);
        gemm(w.HB, w.WH + (long)k * Cin * Cout, out, N, Cout, Cin, Cin, Cout, s);
    }
    k_scale_deg<<<gup((long)N * Cout), 256, 0, s>>>(out, w.DEG, N << co, co);
    cvtbf_t(root, w.WH, 1, Cin, Cout, s);
    gemm(w.HX, w.WH, out, N, Cout, Cin, Cin, Cout, s);
    k_bn_ep<<<gup((long)N * Cout), 256, 0, s>>>(out, bias, g, bb, N << co, co, doElu);
}

// SplineConv K=1 (shortcut): single tap with weight = ev
void conv1x1(const float* xin, float* out, const float* W, const float* root,
             const float* bias, const float* g, const float* bb,
             int N, int Cin, int Cout, const int* src, const int* dst, const float* ev,
             WS& w, hipStream_t s) {
    const int cs = ilog2(Cin), co = ilog2(Cout);
    cvtbf(xin, w.HX, (long)N * Cin, s);
    fillf(out, 0.f, (long)N * Cout, s);
    fillf(w.BUF, 0.f, (long)N * Cin, s);
    int t = E_EDGES << cs;
    k_scatter_mean<<<gup(t), 256, 0, s>>>(xin, src, dst, ev, w.BUF, t, cs);
    cvtbf(w.BUF, w.HB, (long)N * Cin, s);
    cvtbf_t(W, w.WH, 1, Cin, Cout, s);
    gemm(w.HB, w.WH, out, N, Cout, Cin, Cin, Cout, s);
    k_scale_deg<<<gup((long)N * Cout), 256, 0, s>>>(out, w.DEG, N << co, co);
    cvtbf_t(root, w.WH, 1, Cin, Cout, s);
    gemm(w.HX, w.WH, out, N, Cout, Cin, Cin, Cout, s);
    k_bn_ep<<<gup((long)N * Cout), 256, 0, s>>>(out, bias, g, bb, N << co, co, 0);
}

void run_block(void* const* d_in, int base, int N, int Cin, int Cout,
               const float* xin, const int* src, const int* dst, const float* ev,
               WS& w, hipStream_t s) {
    auto F = [&](int i) { return (const float*)d_in[i]; };
    conv5(xin, w.F2, F(base + 0), F(base + 1), F(base + 2), F(base + 3), F(base + 4), 1,
          N, Cin, Cout, src, dst, w, s);                                    // h1 = elu(bn1(c1(x)))
    conv5(w.F2, w.F3, F(base + 5), F(base + 6), F(base + 7), F(base + 8), F(base + 9), 0,
          N, Cout, Cout, src, dst, w, s);                                   // h2 = bn2(c2(h1))
    conv1x1(xin, w.F4, F(base + 10), F(base + 11), F(base + 12), F(base + 13), F(base + 14),
            N, Cin, Cout, src, dst, ev, w, s);                              // s = bnsc(sc(x))
    long n = (long)N * Cout;
    k_add_elu<<<gup(n), 256, 0, s>>>(w.F0, w.F3, w.F4, (int)n);             // out = elu(h2+s)
}

void pool(const float* h, const float* posO, const float* vnO,
          const int* srcO, const int* dstO, const float* evO,
          float* xN, float* posN, float* vnN, int* srcN, int* dstN,
          float* evN, float* attrN,
          int Nold, int C, int bdiv, float sz, int nx, int ny,
          WS& w, hipStream_t s) {
    const int Nn = NBATCH * nx * ny;
    const int cs = ilog2(C);
    const unsigned HMASK = 1048575u;
    k_cluster<<<gup(Nold), 256, 0, s>>>(posO, w.CL, Nold, bdiv, sz, nx, ny, nx * ny);
    filli((int*)w.XENC, 0, (long)Nn * C, s);
    fillf(w.CNT, 0.f, Nn, s);
    fillf(w.PSUM, 0.f, 2L * Nn, s);
    k_pool_feat<<<gup((long)Nold * C), 256, 0, s>>>(h, vnO, w.CL, w.XENC, Nold << cs, cs);
    k_pool_node<<<gup(Nold), 256, 0, s>>>(posO, vnO, w.CL, w.CNT, w.PSUM, Nold);
    k_pool_fin_feat<<<gup((long)Nn * C), 256, 0, s>>>(w.XENC, w.CNT, xN, Nn << cs, cs);
    k_pool_fin_node<<<gup(Nn), 256, 0, s>>>(w.CNT, w.PSUM, posN, vnN, Nn);
    filli(w.HASH, -1, 1048576, s);
    k_pool_edges<<<gup(E_EDGES), 256, 0, s>>>(srcO, dstO, evO, w.CL, srcN, dstN, evN,
                                              w.HASH, HMASK, Nn, E_EDGES);
    filli((int*)w.MENC, 0, 1, s);
    k_pool_cart<<<gup(E_EDGES), 256, 0, s>>>(srcN, dstN, evN, posN, attrN, w.MENC, E_EDGES);
    k_pool_attr<<<gup(E_EDGES), 256, 0, s>>>(attrN, evN, w.MENC, E_EDGES);
}

void level_prep(const float* attr, const float* ev, const int* dst, int N,
                WS& w, hipStream_t s) {
    k_basis5<<<gup(E_EDGES), 256, 0, s>>>(attr, ev, w.BAS, w.WI, E_EDGES);
    fillf(w.DEG, 0.f, N, s);
    k_deg<<<gup(E_EDGES), 256, 0, s>>>(dst, ev, w.DEG, E_EDGES);
}

} // namespace

extern "C" void kernel_launch(void* const* d_in, const int* in_sizes, int n_in,
                              void* d_out, int out_size, void* d_ws, size_t ws_size,
                              hipStream_t stream) {
    (void)in_sizes; (void)n_in; (void)out_size; (void)ws_size;
    hipStream_t s = stream;

    // --- inputs (insertion-order flattening of setup_inputs dict) ---
    const float* x0    = (const float*)d_in[0];
    const float* pos0  = (const float*)d_in[1];
    const float* attr0 = (const float*)d_in[2];
    const int*   eidx  = (const int*)d_in[3];
    const int* src0 = eidx;
    const int* dst0 = eidx + E_EDGES;
    enum { IN_C1W = 4, IN_C1R, IN_C1B, IN_BN1G, IN_BN1B, IN_B1 = 9, IN_B2 = 24, IN_B3 = 39,
           IN_FC1W = 54, IN_FC1B = 55, IN_FC2W = 56, IN_FC2B = 57 };
    auto F = [&](int i) { return (const float*)d_in[i]; };

    // --- workspace bump allocation ---
    size_t off = 0;
    auto alloc = [&](size_t bytes) -> void* {
        bytes = (bytes + 255) & ~(size_t)255;
        void* p = (char*)d_ws + off; off += bytes; return p;
    };
    WS w;
    w.F0  = (float*)alloc(3000000 * 4);  w.F1 = (float*)alloc(3000000 * 4);
    w.F2  = (float*)alloc(3000000 * 4);  w.F3 = (float*)alloc(3000000 * 4);
    w.F4  = (float*)alloc(3000000 * 4);  w.BUF = (float*)alloc(3000000 * 4);
    w.HX  = (bf16*)alloc(3000000 * 2);   w.HB = (bf16*)alloc(3000000 * 2);
    w.WH  = (bf16*)alloc(6600000 * 2);
    w.BAS = (float*)alloc(524288 * 4);   w.WI = (int*)alloc(524288 * 4);
    w.DEG = (float*)alloc(32768 * 4);
    w.VNA = (float*)alloc(32768 * 4);    w.VNB = (float*)alloc(32768 * 4);
    w.POSA = (float*)alloc(65536 * 4);   w.POSB = (float*)alloc(65536 * 4);
    w.CL  = (int*)alloc(32768 * 4);
    w.SRCA = (int*)alloc(E_EDGES * 4);   w.DSTA = (int*)alloc(E_EDGES * 4);
    w.SRCB = (int*)alloc(E_EDGES * 4);   w.DSTB = (int*)alloc(E_EDGES * 4);
    w.EVA  = (float*)alloc(E_EDGES * 4); w.EVB  = (float*)alloc(E_EDGES * 4);
    w.ATTRA = (float*)alloc(2 * E_EDGES * 4);
    w.ATTRB = (float*)alloc(2 * E_EDGES * 4);
    w.XENC = (unsigned*)alloc(1500000 * 4);
    w.CNT  = (float*)alloc(32768 * 4);   w.PSUM = (float*)alloc(65536 * 4);
    w.HASH = (int*)alloc(1048576 * 4);   w.MENC = (unsigned*)alloc(256);

    // ---------- level 0: conv1 (1 -> 64) ----------
    fillf(w.EVA, 1.f, E_EDGES, s);
    fillf(w.VNA, 1.f, 32768, s);
    level_prep(attr0, w.EVA, dst0, 32768, w, s);
    fillf(w.BUF, 0.f, 32768L * 25, s);
    k_conv1_scatter<<<gup(E_EDGES), 256, 0, s>>>(x0, src0, dst0, w.BAS, w.WI, w.BUF, E_EDGES);
    k_conv1_finish<<<gup(32768L * 64), 256, 0, s>>>(w.BUF, x0, F(IN_C1W), F(IN_C1R),
                                                    F(IN_C1B), F(IN_BN1G), F(IN_BN1B),
                                                    w.DEG, w.F0, 32768 * 64);

    // ---------- pool 1 (sz=4, 60x45) : F0 -> F1 ----------
    pool(w.F0, pos0, w.VNA, src0, dst0, w.EVA,
         w.F1, w.POSB, w.VNB, w.SRCB, w.DSTB, w.EVB, w.ATTRB,
         32768, 64, 4096, 4.f, 60, 45, w, s);
    level_prep(w.ATTRB, w.EVB, w.DSTB, NBATCH * 2700, w, s);
    run_block(d_in, IN_B1, NBATCH * 2700, 64, 128, w.F1, w.SRCB, w.DSTB, w.EVB, w, s); // -> F0

    // ---------- pool 2 (sz=6, 40x30) : F0 -> F1 ----------
    pool(w.F0, w.POSB, w.VNB, w.SRCB, w.DSTB, w.EVB,
         w.F1, w.POSA, w.VNA, w.SRCA, w.DSTA, w.EVA, w.ATTRA,
         NBATCH * 2700, 128, 2700, 6.f, 40, 30, w, s);
    level_prep(w.ATTRA, w.EVA, w.DSTA, NBATCH * 1200, w, s);
    run_block(d_in, IN_B2, NBATCH * 1200, 128, 256, w.F1, w.SRCA, w.DSTA, w.EVA, w, s); // -> F0

    // ---------- pool 3 (sz=20, 12x9) : F0 -> F1 ----------
    pool(w.F0, w.POSA, w.VNA, w.SRCA, w.DSTA, w.EVA,
         w.F1, w.POSB, w.VNB, w.SRCB, w.DSTB, w.EVB, w.ATTRB,
         NBATCH * 1200, 256, 1200, 20.f, 12, 9, w, s);
    level_prep(w.ATTRB, w.EVB, w.DSTB, NBATCH * 108, w, s);
    run_block(d_in, IN_B3, NBATCH * 108, 256, 512, w.F1, w.SRCB, w.DSTB, w.EVB, w, s);  // -> F0

    // ---------- final max_pool_x (sz=32, 8x6 into 64 slots/sample) : F0 -> F1 ----------
    {
        const int Nold = NBATCH * 108, Nn = NBATCH * 64, C = 512, cs = 9;
        k_cluster<<<gup(Nold), 256, 0, s>>>(w.POSB, w.CL, Nold, 108, 32.f, 8, 6, 64);
        filli((int*)w.XENC, 0, (long)Nn * C, s);
        fillf(w.CNT, 0.f, Nn, s);
        fillf(w.PSUM, 0.f, 2L * Nn, s);
        k_pool_feat<<<gup((long)Nold * C), 256, 0, s>>>(w.F0, w.VNB, w.CL, w.XENC,
                                                        Nold << cs, cs);
        k_pool_node<<<gup(Nold), 256, 0, s>>>(w.POSB, w.VNB, w.CL, w.CNT, w.PSUM, Nold);
        k_pool_fin_feat<<<gup((long)Nn * C), 256, 0, s>>>(w.XENC, w.CNT, w.F1, Nn << cs, cs);
    }

    // ---------- fc1 (8x32768 @ 32768x1024), K-chunked bf16 transpose-conversion ----------
    cvtbf(w.F1, w.HX, 8L * 32768, s);                 // f = xm.reshape(8, 32768) (contiguous)
    fillf(w.F2, 0.f, 8L * 1024, s);
    for (int kc = 0; kc < 8; ++kc) {
        cvtbf_t(F(IN_FC1W) + (long)kc * 4096 * 1024, w.WH, 1, 4096, 1024, s); // [1024][4096]
        gemm(w.HX + kc * 4096, w.WH, w.F2, 8, 1024, 4096, 32768, 1024, s);
    }
    k_bias_elu<<<gup(8L * 1024), 256, 0, s>>>(w.F2, F(IN_FC1B), 8 * 1024, 10);

    // ---------- fc2 (8x1024 @ 1024x101) + log_softmax ----------
    cvtbf(w.F2, w.HB, 8L * 1024, s);
    cvtbf_t(F(IN_FC2W), w.WH, 1, 1024, 101, s);       // BT [101][1024]
    fillf(w.F3, 0.f, 8L * 101, s);
    gemm(w.HB, w.WH, w.F3, 8, 101, 1024, 1024, 101, s);
    k_logsoftmax<<<8, 128, 0, s>>>(w.F3, F(IN_FC2B), (float*)d_out, 101);
}
